// DiceCoefficientAll_32384053412395
// MI455X (gfx1250) — compile-verified, matrix operand order checked by reading, not run on hardware
//
#include <hip/hip_runtime.h>
#include <hip/hip_bf16.h>

// CDNA5 / gfx1250: wave32, WMMA 16x16x32 f16.
typedef __attribute__((ext_vector_type(16))) _Float16 v16h;
typedef __attribute__((ext_vector_type(8)))  float    v8f;

#define NUM_LABELS 14
#define PAD_BINS   48          // 3 WMMA tiles of 16 columns
#define COPIES     32          // one histogram copy per lane id -> conflict-free ds_add
#define BLK        256         // 8 wave32 waves
#define CHUNK      16384       // elements per block (per input)

// d_ws accumulator layout: float accum[8 batches][3 regions][48 bins]
// region 0 = count1, 1 = count2, 2 = inter
#define ACC_FLOATS (8 * 3 * PAD_BINS)

__global__ void dice_zero_kernel(float* __restrict__ accum, int n) {
    int i = blockIdx.x * blockDim.x + threadIdx.x;
    if (i < n) accum[i] = 0.0f;
}

__global__ __launch_bounds__(BLK)
void dice_hist_kernel(const float* __restrict__ s1, const float* __restrict__ s2,
                      float* __restrict__ accum, int N, int BPB) {
    // 3 regions x 48 bins x 32 copies of u32 = 18 KB LDS
    __shared__ unsigned hist[3][PAD_BINS * COPIES];

    unsigned* hflat = &hist[0][0];
    for (int i = threadIdx.x; i < 3 * PAD_BINS * COPIES; i += BLK) hflat[i] = 0u;
    __syncthreads();

    const int batch = blockIdx.x / BPB;
    const int blk   = blockIdx.x % BPB;
    const long base = (long)batch * (long)N + (long)blk * (long)CHUNK;
    const int  copy = threadIdx.x & 31;           // lane id (wave32)

    const float4* __restrict__ p1 = (const float4*)(s1 + base);
    const float4* __restrict__ p2 = (const float4*)(s2 + base);
    const int nvec = CHUNK >> 2;                  // 4096 float4 per block

    // Streaming pass: 2x b128 loads + 8..12 conflict-free ds_add_u32 per float4.
    for (int i = threadIdx.x; i < nvec; i += BLK) {
        float4 a = p1[i];
        float4 b = p2[i];
        int a0 = (int)a.x, a1 = (int)a.y, a2 = (int)a.z, a3 = (int)a.w;
        int b0 = (int)b.x, b1 = (int)b.y, b2 = (int)b.z, b3 = (int)b.w;

        atomicAdd(&hist[0][a0 * COPIES + copy], 1u);
        atomicAdd(&hist[0][a1 * COPIES + copy], 1u);
        atomicAdd(&hist[0][a2 * COPIES + copy], 1u);
        atomicAdd(&hist[0][a3 * COPIES + copy], 1u);

        atomicAdd(&hist[1][b0 * COPIES + copy], 1u);
        atomicAdd(&hist[1][b1 * COPIES + copy], 1u);
        atomicAdd(&hist[1][b2 * COPIES + copy], 1u);
        atomicAdd(&hist[1][b3 * COPIES + copy], 1u);

        if (a0 == b0) atomicAdd(&hist[2][a0 * COPIES + copy], 1u);
        if (a1 == b1) atomicAdd(&hist[2][a1 * COPIES + copy], 1u);
        if (a2 == b2) atomicAdd(&hist[2][a2 * COPIES + copy], 1u);
        if (a3 == b3) atomicAdd(&hist[2][a3 * COPIES + copy], 1u);
    }
    __syncthreads();

    // Copy reduction as a matrix op: colsum(B[32x16]) = ones(16x32) x B.
    // D[m][n] = sum_k B[k][n] for every m, so d[0] on each lane is the
    // column sum for bin = tile*16 + (lane&15).
    const int wave = threadIdx.x >> 5;
    const int lane = threadIdx.x & 31;

    v16h ones;
#pragma unroll
    for (int k = 0; k < 16; ++k) ones[k] = (_Float16)1.0f;

    // 9 tasks = 3 regions x 3 tiles, spread over the 8 waves (branch uniform
    // per wave -> EXEC all ones for the WMMA, as required).
    for (int task = wave; task < 9; task += (BLK >> 5)) {
        const int region = task / 3;
        const int tile   = task % 3;
        const int n      = lane & 15;            // column (bin within tile)
        const int kbase  = lane & 16;            // copies 0..15 or 16..31

        v16h bm;
#pragma unroll
        for (int k = 0; k < 16; ++k) {
            unsigned cnt = hist[region][(tile * 16 + n) * COPIES + kbase + k];
            bm[k] = (_Float16)(float)cnt;        // exact: cnt <= 512 < 2048
        }

        v8f c = {};
        v8f d = __builtin_amdgcn_wmma_f32_16x16x32_f16(
            /*neg_a=*/false, ones, /*neg_b=*/false, bm,
            /*c_mod=*/(short)0, c, /*reuse_a=*/false, /*reuse_b=*/false);

        if (lane < 16) {  // lanes 16..31 hold duplicate column sums
            atomicAdd(&accum[(batch * 3 + region) * PAD_BINS + tile * 16 + n], d[0]);
        }
    }
}

__global__ void dice_finalize_kernel(const float* __restrict__ accum,
                                     float* __restrict__ out) {
    __shared__ float dice[8 * NUM_LABELS];
    const int t = threadIdx.x;
    if (t < 8 * NUM_LABELS) {
        const int b = t / NUM_LABELS;
        const int l = t % NUM_LABELS;
        float c1 = accum[(b * 3 + 0) * PAD_BINS + l];
        float c2 = accum[(b * 3 + 1) * PAD_BINS + l];
        float it = accum[(b * 3 + 2) * PAD_BINS + l];
        dice[t] = 2.0f * it / (c1 + c2 + 2.2204460492503131e-16f);
    }
    __syncthreads();
    if (t == 0) {
        float total = 0.0f;
        int nz = 0;
        for (int i = 0; i < 8 * NUM_LABELS; ++i) {
            total += dice[i];
            if (dice[i] > 0.0f) ++nz;
        }
        out[0] = (nz > 0) ? (total / (float)nz) : 0.0f;
    }
}

extern "C" void kernel_launch(void* const* d_in, const int* in_sizes, int n_in,
                              void* d_out, int out_size, void* d_ws, size_t ws_size,
                              hipStream_t stream) {
    const float* s1 = (const float*)d_in[0];
    const float* s2 = (const float*)d_in[1];
    float* accum = (float*)d_ws;          // 1152 floats of scratch
    float* out   = (float*)d_out;

    const int total = in_sizes[0];        // 8 * 128^3 = 16,777,216
    const int B = 8;
    const int N = total / B;              // 2,097,152 per batch
    const int BPB = N / CHUNK;            // 128 blocks per batch (exact)

    dice_zero_kernel<<<(ACC_FLOATS + 255) / 256, 256, 0, stream>>>(accum, ACC_FLOATS);
    dice_hist_kernel<<<B * BPB, BLK, 0, stream>>>(s1, s2, accum, N, BPB);
    dice_finalize_kernel<<<1, 128, 0, stream>>>(accum, out);
}
